// GTM_15702400434566
// MI455X (gfx1250) — compile-verified, matrix operand order checked by reading, not run on hardware
//
#include <hip/hip_runtime.h>

#define Bc 8
#define Lc 2048
#define INDIM 1024
#define Dc 64
#define Hc 2
#define DHc 32
#define NNc 32
#define LNEPS 1e-5f

typedef _Float16 v8h  __attribute__((ext_vector_type(8)));
typedef _Float16 v16h __attribute__((ext_vector_type(16)));
typedef float    v8f  __attribute__((ext_vector_type(8)));

// ---------------------------------------------------------------------------
// Kernel A: LN(1024) -> GEMM(1024x64) -> leaky -> LN(64) -> GEMM(64x64)
//           -> leaky -> LN(64).  One block (256 threads) per row.
// ---------------------------------------------------------------------------
__global__ void __launch_bounds__(256) front_mlp_kernel(
    const float* __restrict__ xin,
    const float* __restrict__ lnig, const float* __restrict__ lnib,
    const float* __restrict__ w_in, const float* __restrict__ b_in,
    const float* __restrict__ ln1g, const float* __restrict__ ln1b,
    const float* __restrict__ w_h,  const float* __restrict__ b_h,
    const float* __restrict__ ln2g, const float* __restrict__ ln2b,
    float* __restrict__ X, _Float16* __restrict__ Xh)
{
  __shared__ float sx[INDIM];
  __shared__ float red[256];
  __shared__ float part[4][Dc];
  __shared__ float s2[Dc];
  const int row = blockIdx.x;              // b*L + l
  const int t   = threadIdx.x;
  const float* xr = xin + (size_t)row * INDIM;

  float4 v = reinterpret_cast<const float4*>(xr)[t];
  red[t] = v.x + v.y + v.z + v.w;
  __syncthreads();
  for (int s = 128; s > 0; s >>= 1) { if (t < s) red[t] += red[t + s]; __syncthreads(); }
  float mean = red[0] * (1.0f / INDIM);
  __syncthreads();
  float dx0 = v.x - mean, dx1 = v.y - mean, dx2 = v.z - mean, dx3 = v.w - mean;
  red[t] = dx0*dx0 + dx1*dx1 + dx2*dx2 + dx3*dx3;
  __syncthreads();
  for (int s = 128; s > 0; s >>= 1) { if (t < s) red[t] += red[t + s]; __syncthreads(); }
  float rs = rsqrtf(red[0] * (1.0f / INDIM) + LNEPS);
  __syncthreads();
  int k0 = t * 4;
  sx[k0+0] = dx0 * rs * lnig[k0+0] + lnib[k0+0];
  sx[k0+1] = dx1 * rs * lnig[k0+1] + lnib[k0+1];
  sx[k0+2] = dx2 * rs * lnig[k0+2] + lnib[k0+2];
  sx[k0+3] = dx3 * rs * lnig[k0+3] + lnib[k0+3];
  __syncthreads();

  // GEMM1: 1024 -> 64 (4 k-chunks x 64 outputs)
  const int n  = t & 63;
  const int kc = t >> 6;
  float acc = 0.f;
  const int kbeg = kc * 256;
  for (int k = kbeg; k < kbeg + 256; ++k)
    acc = fmaf(sx[k], w_in[(size_t)k * Dc + n], acc);
  part[kc][n] = acc;
  __syncthreads();
  float z1 = 0.f;
  if (t < Dc) {
    z1 = part[0][t] + part[1][t] + part[2][t] + part[3][t] + b_in[t];
    z1 = z1 > 0.f ? z1 : 0.01f * z1;
  }
  // LN over 64
  red[t] = (t < Dc) ? z1 : 0.f;
  __syncthreads();
  for (int s = 128; s > 0; s >>= 1) { if (t < s) red[t] += red[t + s]; __syncthreads(); }
  float m1 = red[0] * (1.0f / Dc);
  __syncthreads();
  float d1 = (t < Dc) ? (z1 - m1) : 0.f;
  red[t] = d1 * d1;
  __syncthreads();
  for (int s = 128; s > 0; s >>= 1) { if (t < s) red[t] += red[t + s]; __syncthreads(); }
  float rsA = rsqrtf(red[0] * (1.0f / Dc) + LNEPS);
  __syncthreads();
  if (t < Dc) s2[t] = d1 * rsA * ln1g[t] + ln1b[t];
  __syncthreads();

  // GEMM2: 64 -> 64
  float z2 = 0.f;
  if (t < Dc) {
    #pragma unroll 8
    for (int k = 0; k < Dc; ++k) z2 = fmaf(s2[k], w_h[(size_t)k * Dc + t], z2);
    z2 += b_h[t];
    z2 = z2 > 0.f ? z2 : 0.01f * z2;
  }
  red[t] = (t < Dc) ? z2 : 0.f;
  __syncthreads();
  for (int s = 128; s > 0; s >>= 1) { if (t < s) red[t] += red[t + s]; __syncthreads(); }
  float m2 = red[0] * (1.0f / Dc);
  __syncthreads();
  float d2 = (t < Dc) ? (z2 - m2) : 0.f;
  red[t] = d2 * d2;
  __syncthreads();
  for (int s = 128; s > 0; s >>= 1) { if (t < s) red[t] += red[t + s]; __syncthreads(); }
  float rsB = rsqrtf(red[0] * (1.0f / Dc) + LNEPS);
  __syncthreads();
  if (t < Dc) {
    float f = d2 * rsB * ln2g[t] + ln2b[t];
    X[(size_t)row * Dc + t] = f;
    int b = row >> 11, l = row & (Lc - 1);
    int head = t >> 5, kk = t & 31;
    Xh[(((size_t)(b * Hc + head)) * Lc + l) * DHc + kk] = (_Float16)f;
  }
}

// ---------------------------------------------------------------------------
// Kernel T: per (b,i) select top-32 columns of mask_j * rsqrt(1+dist_ij),
// ties broken toward lower index (stable-argsort semantics).
// ---------------------------------------------------------------------------
__global__ void __launch_bounds__(256) top32_kernel(
    const float* __restrict__ dist, const float* __restrict__ masks,
    int* __restrict__ topi)
{
  __shared__ float dw[Lc];
  __shared__ float bv[256];
  __shared__ int   bi[256];
  __shared__ float s_pv;
  __shared__ int   s_pj;
  const int row = blockIdx.x;       // b*L + i
  const int b   = row >> 11;
  const int t   = threadIdx.x;
  const float* dr = dist + (size_t)row * Lc;
  for (int j = t; j < Lc; j += 256)
    dw[j] = masks[(size_t)b * Lc + j] * rsqrtf(1.0f + dr[j]);
  if (t == 0) { s_pv = __builtin_inff(); s_pj = -1; }
  __syncthreads();
  for (int it = 0; it < NNc; ++it) {
    float pv = s_pv; int pj = s_pj;
    float best = -__builtin_inff(); int besti = 1 << 30;
    for (int j = t; j < Lc; j += 256) {
      float vv = dw[j];
      bool elig = (vv < pv) || (vv == pv && j > pj);
      if (elig && (vv > best || (vv == best && j < besti))) { best = vv; besti = j; }
    }
    bv[t] = best; bi[t] = besti;
    __syncthreads();
    for (int s = 128; s > 0; s >>= 1) {
      if (t < s) {
        float ov = bv[t + s]; int oi = bi[t + s];
        if (ov > bv[t] || (ov == bv[t] && oi < bi[t])) { bv[t] = ov; bi[t] = oi; }
      }
      __syncthreads();
    }
    if (t == 0) { topi[(size_t)row * NNc + it] = bi[0]; s_pv = bv[0]; s_pj = bi[0]; }
    __syncthreads();
  }
}

// ---------------------------------------------------------------------------
// Kernel C: two-pass softmax row stats (max, Z) of scores = h_i.h_j + add_mask.
// One wave owns 32 rows (two A fragments); pass 1 = WMMA + fmax only,
// pass 2 = WMMA + single exp per element.  2 WMMAs per B fetch.
// ---------------------------------------------------------------------------
__global__ void __launch_bounds__(256) attn_stats_kernel(
    const _Float16* __restrict__ Xh, const float* __restrict__ masks,
    float* __restrict__ Mst, float* __restrict__ Zst)
{
  const int lane  = threadIdx.x & 31;
  const int wave  = threadIdx.x >> 5;
  const int group = blockIdx.x * 8 + wave;   // 0 .. B*H*L/32 - 1 (1024)
  const int bh    = group >> 6;              // 64 32-row groups per (b,h)
  const int row0  = (group & 63) << 5;
  const int b     = bh >> 1;
  const _Float16* base = Xh + (size_t)bh * Lc * DHc;

  const int klo = (lane < 16) ? 0 : 8;
  const int mr0 = row0 + (lane & 15);
  const int mr1 = mr0 + 16;
  v16h A0, A1;
  {
    v8h lo = *reinterpret_cast<const v8h*>(base + (size_t)mr0 * DHc + klo);
    v8h hi = *reinterpret_cast<const v8h*>(base + (size_t)mr0 * DHc + 16 + klo);
    #pragma unroll
    for (int i = 0; i < 8; ++i) { A0[i] = lo[i]; A0[8 + i] = hi[i]; }
    lo = *reinterpret_cast<const v8h*>(base + (size_t)mr1 * DHc + klo);
    hi = *reinterpret_cast<const v8h*>(base + (size_t)mr1 * DHc + 16 + klo);
    #pragma unroll
    for (int i = 0; i < 8; ++i) { A1[i] = lo[i]; A1[8 + i] = hi[i]; }
  }

  float mx[16];
  #pragma unroll
  for (int r = 0; r < 16; ++r) mx[r] = -3.0e38f;

  // ---- pass 1: row maxima (no transcendentals) ----
  for (int ct = 0; ct < Lc / 16; ++ct) {
    const int col = ct * 16 + (lane & 15);
    v8h blo = *reinterpret_cast<const v8h*>(base + (size_t)col * DHc + klo);
    v8h bhi = *reinterpret_cast<const v8h*>(base + (size_t)col * DHc + 16 + klo);
    v16h Bt;
    #pragma unroll
    for (int i = 0; i < 8; ++i) { Bt[i] = blo[i]; Bt[8 + i] = bhi[i]; }
    float am = (1.0f - masks[(size_t)b * Lc + col]) * -10000.0f;
    v8f C0 = {};
    C0 = __builtin_amdgcn_wmma_f32_16x16x32_f16(false, A0, false, Bt,
                                                (short)0, C0, false, false);
    #pragma unroll
    for (int r = 0; r < 8; ++r) mx[r] = fmaxf(mx[r], C0[r] + am);
    v8f C1 = {};
    C1 = __builtin_amdgcn_wmma_f32_16x16x32_f16(false, A1, false, Bt,
                                                (short)0, C1, false, false);
    #pragma unroll
    for (int r = 0; r < 8; ++r) mx[8 + r] = fmaxf(mx[8 + r], C1[r] + am);
  }
  // merge maxima across the 16 lanes of each half (columns)
  #pragma unroll
  for (int off = 1; off < 16; off <<= 1) {
    #pragma unroll
    for (int r = 0; r < 16; ++r)
      mx[r] = fmaxf(mx[r], __shfl_xor(mx[r], off, 32));
  }

  // ---- pass 2: Z = sum exp(s - max) (one exp per element) ----
  float sm[16];
  #pragma unroll
  for (int r = 0; r < 16; ++r) sm[r] = 0.f;
  for (int ct = 0; ct < Lc / 16; ++ct) {
    const int col = ct * 16 + (lane & 15);
    v8h blo = *reinterpret_cast<const v8h*>(base + (size_t)col * DHc + klo);
    v8h bhi = *reinterpret_cast<const v8h*>(base + (size_t)col * DHc + 16 + klo);
    v16h Bt;
    #pragma unroll
    for (int i = 0; i < 8; ++i) { Bt[i] = blo[i]; Bt[8 + i] = bhi[i]; }
    float am = (1.0f - masks[(size_t)b * Lc + col]) * -10000.0f;
    v8f C0 = {};
    C0 = __builtin_amdgcn_wmma_f32_16x16x32_f16(false, A0, false, Bt,
                                                (short)0, C0, false, false);
    #pragma unroll
    for (int r = 0; r < 8; ++r) sm[r] += __expf(C0[r] + am - mx[r]);
    v8f C1 = {};
    C1 = __builtin_amdgcn_wmma_f32_16x16x32_f16(false, A1, false, Bt,
                                                (short)0, C1, false, false);
    #pragma unroll
    for (int r = 0; r < 8; ++r) sm[8 + r] += __expf(C1[r] + am - mx[8 + r]);
  }
  // sum-reduce across the 16 lanes of each half
  #pragma unroll
  for (int off = 1; off < 16; off <<= 1) {
    #pragma unroll
    for (int r = 0; r < 16; ++r)
      sm[r] += __shfl_xor(sm[r], off, 32);
  }

  if ((lane & 15) == 0) {
    const int half = lane >> 4;     // 0: rows r / r+16 ; 1: rows r+8 / r+24
    #pragma unroll
    for (int r = 0; r < 8; ++r) {
      int ra = row0 + half * 8 + r;          // from A0's C tile
      Mst[(size_t)bh * Lc + ra] = mx[r];
      Zst[(size_t)bh * Lc + ra] = sm[r];
      int rb = ra + 16;                      // from A1's C tile
      Mst[(size_t)bh * Lc + rb] = mx[8 + r];
      Zst[(size_t)bh * Lc + rb] = sm[8 + r];
    }
  }
}

// ---------------------------------------------------------------------------
// Kernel D: gather top-32 columns, exact f32 scores, renormalized weighted sum
// out_i = sum_j e_ij h_j / (S_top + 1e-5*Z), then LayerNorm(64).
// One block of 64 threads per row (thread t = output dim, head = t>>5).
// ---------------------------------------------------------------------------
__global__ void __launch_bounds__(64) attn_apply_kernel(
    const float* __restrict__ Xf, const float* __restrict__ masks,
    const int* __restrict__ topi,
    const float* __restrict__ Mst, const float* __restrict__ Zst,
    const float* __restrict__ lng, const float* __restrict__ lnb,
    float* __restrict__ Yf, _Float16* __restrict__ Yh)
{
  __shared__ float si[Dc];
  __shared__ int   tj[NNc];
  __shared__ float ev[Dc];
  __shared__ float red[Dc];
  const int row  = blockIdx.x;        // b*L + i
  const int b    = row >> 11;
  const int i    = row & (Lc - 1);
  const int t    = threadIdx.x;       // 0..63
  const int head = t >> 5;
  const int slot = t & 31;

  si[t] = Xf[(size_t)row * Dc + t];
  if (t < NNc) tj[t] = topi[(size_t)row * NNc + t];
  __syncthreads();

  const int j = tj[slot];
  const float* hj = Xf + ((size_t)b * Lc + j) * Dc + head * DHc;
  float s = 0.f;
  #pragma unroll
  for (int k = 0; k < DHc; ++k) s = fmaf(si[head * DHc + k], hj[k], s);
  s += (1.0f - masks[(size_t)b * Lc + j]) * -10000.0f;
  const size_t sidx = ((size_t)(b * Hc + head)) * Lc + i;
  float m = Mst[sidx];
  float Z = Zst[sidx];
  ev[t] = __expf(s - m);
  __syncthreads();

  float St = 0.f;
  #pragma unroll 8
  for (int q = 0; q < NNc; ++q) St += ev[head * NNc + q];
  float inv = 1.0f / (St + 1e-5f * Z);

  float out = 0.f;
  for (int q = 0; q < NNc; ++q) {
    int jq = tj[q];
    out = fmaf(ev[head * NNc + q], Xf[((size_t)b * Lc + jq) * Dc + t], out);
  }
  out *= inv;

  // LayerNorm over 64
  red[t] = out;
  __syncthreads();
  for (int sft = 32; sft > 0; sft >>= 1) { if (t < sft) red[t] += red[t + sft]; __syncthreads(); }
  float mean = red[0] * (1.0f / Dc);
  __syncthreads();
  float d = out - mean;
  red[t] = d * d;
  __syncthreads();
  for (int sft = 32; sft > 0; sft >>= 1) { if (t < sft) red[t] += red[t + sft]; __syncthreads(); }
  float rs = rsqrtf(red[0] * (1.0f / Dc) + LNEPS);
  __syncthreads();
  float f = d * rs * lng[t] + lnb[t];
  Yf[(size_t)row * Dc + t] = f;
  Yh[(((size_t)(b * Hc + head)) * Lc + i) * DHc + slot] = (_Float16)f;
}

// ---------------------------------------------------------------------------
// Kernel E: y = X @ w_out + b_out
// ---------------------------------------------------------------------------
__global__ void __launch_bounds__(256) out_proj_kernel(
    const float* __restrict__ Xf, const float* __restrict__ w_out,
    const float* __restrict__ b_out, float* __restrict__ y)
{
  int r = blockIdx.x * blockDim.x + threadIdx.x;
  if (r >= Bc * Lc) return;
  const float* xr = Xf + (size_t)r * Dc;
  float acc = b_out[0];
  #pragma unroll 8
  for (int d = 0; d < Dc; ++d) acc = fmaf(xr[d], w_out[d], acc);
  y[r] = acc;
}

// ---------------------------------------------------------------------------
extern "C" void kernel_launch(void* const* d_in, const int* in_sizes, int n_in,
                              void* d_out, int out_size, void* d_ws, size_t ws_size,
                              hipStream_t stream)
{
  (void)in_sizes; (void)n_in; (void)out_size; (void)ws_size;
  const float* xin    = (const float*)d_in[0];
  // d_in[1] protein_edge_features: unused by the reference
  const float* dist   = (const float*)d_in[2];
  const float* masks  = (const float*)d_in[3];
  const float* ln_in_g= (const float*)d_in[4];
  const float* ln_in_b= (const float*)d_in[5];
  const float* w_in   = (const float*)d_in[6];
  const float* b_in   = (const float*)d_in[7];
  const float* ln1g   = (const float*)d_in[8];
  const float* ln1b   = (const float*)d_in[9];
  const float* w_h    = (const float*)d_in[10];
  const float* b_h    = (const float*)d_in[11];
  const float* ln2g   = (const float*)d_in[12];
  const float* ln2b   = (const float*)d_in[13];
  const float* ln_a0g = (const float*)d_in[14];
  const float* ln_a0b = (const float*)d_in[15];
  const float* ln_a1g = (const float*)d_in[16];
  const float* ln_a1b = (const float*)d_in[17];
  const float* w_out  = (const float*)d_in[18];
  const float* b_out  = (const float*)d_in[19];
  float* y = (float*)d_out;

  char* w = (char*)d_ws;
  size_t off = 0;
  auto carve = [&](size_t bytes) -> void* {
    void* p = (void*)(w + off);
    off += (bytes + 255) & ~(size_t)255;
    return p;
  };
  float*    X   = (float*)   carve((size_t)Bc * Lc * Dc * sizeof(float));   // 4 MB
  float*    Y   = (float*)   carve((size_t)Bc * Lc * Dc * sizeof(float));   // 4 MB
  _Float16* Xh  = (_Float16*)carve((size_t)Bc * Hc * Lc * DHc * 2);         // 2 MB
  _Float16* Yh  = (_Float16*)carve((size_t)Bc * Hc * Lc * DHc * 2);         // 2 MB
  float*    Mst = (float*)   carve((size_t)Bc * Hc * Lc * sizeof(float));   // 128 KB
  float*    Zst = (float*)   carve((size_t)Bc * Hc * Lc * sizeof(float));   // 128 KB
  int*      top = (int*)     carve((size_t)Bc * Lc * NNc * sizeof(int));    // 2 MB

  const int ROWS = Bc * Lc;                 // 16384

  // Front MLP
  front_mlp_kernel<<<ROWS, 256, 0, stream>>>(
      xin, ln_in_g, ln_in_b, w_in, b_in, ln1g, ln1b, w_h, b_h, ln2g, ln2b, X, Xh);

  // Top-32 selection from dist weights (score-independent)
  top32_kernel<<<ROWS, 256, 0, stream>>>(dist, masks, top);

  const int statBlocks = (Bc * Hc * Lc / 32) / 8;   // 1024 waves / 8 per block = 128

  // Attention layer 1
  attn_stats_kernel<<<statBlocks, 256, 0, stream>>>(Xh, masks, Mst, Zst);
  attn_apply_kernel<<<ROWS, 64, 0, stream>>>(X, masks, top, Mst, Zst,
                                             ln_a0g, ln_a0b, Y, Yh);
  // Attention layer 2
  attn_stats_kernel<<<statBlocks, 256, 0, stream>>>(Yh, masks, Mst, Zst);
  attn_apply_kernel<<<ROWS, 64, 0, stream>>>(Y, masks, top, Mst, Zst,
                                             ln_a1g, ln_a1b, X, Xh);

  // Output projection
  out_proj_kernel<<<(ROWS + 255) / 256, 256, 0, stream>>>(X, w_out, b_out, y);
}
